// JointSRLModule_45492293599337
// MI455X (gfx1250) — compile-verified, matrix operand order checked by reading, not run on hardware
//
#include <hip/hip_runtime.h>
#include <hip/hip_bf16.h>

// ---------------- problem constants (from reference) ----------------
#define BB    8
#define TT    512
#define HH    1024
#define NA    1024
#define NP    512
#define NWW   16
#define WDD   20
#define NCLS  66
#define KARG  30
#define KPRED 15
#define ARG_D  3092          // 3H + WD
#define ARG_DP 3104          // padded to 32 for WMMA K loop
#define PRED_D 2048
#define NEGV  (-1e20f)

typedef __bf16 bf16_t;
typedef __attribute__((ext_vector_type(8)))  bf16_t v8bf;
typedef __attribute__((ext_vector_type(16))) bf16_t v16bf;
typedef __attribute__((ext_vector_type(8)))  float  v8f;

static __device__ __forceinline__ bf16_t f2bf(float f) {
  unsigned u = __builtin_bit_cast(unsigned, f);
  unsigned r = u + 0x7FFFu + ((u >> 16) & 1u);   // round-to-nearest-even
  unsigned short hs = (unsigned short)(r >> 16);
  return __builtin_bit_cast(bf16_t, hs);
}

static __device__ __forceinline__ v16bf load_frag(const bf16_t* p) {
  v8bf lo = *(const v8bf*)(p);
  v8bf hi = *(const v8bf*)(p + 16);
  return __builtin_shufflevector(lo, hi, 0, 1, 2, 3, 4, 5, 6, 7,
                                 8, 9, 10, 11, 12, 13, 14, 15);
}

// ---------------- weight transpose + f32->bf16 convert ----------------
// in: [K, N] f32 row-major. out: [Npad, Kpad] bf16, out[n*Kpad+k] = in[k*N+n], 0 pad.
__global__ void transpose_cvt_kernel(const float* __restrict__ in, int K, int N,
                                     bf16_t* __restrict__ out, int Kpad, int Npad) {
  size_t gid = (size_t)blockIdx.x * blockDim.x + threadIdx.x;
  size_t tot = (size_t)Npad * Kpad;
  if (gid >= tot) return;
  int k = (int)(gid % Kpad);
  int n = (int)(gid / Kpad);
  float v = (k < K && n < N) ? in[(size_t)k * N + n] : 0.f;
  out[gid] = f2bf(v);
}

__global__ void pad_bias_kernel(const float* __restrict__ bs2, float* __restrict__ out) {
  int i = threadIdx.x;
  if (i < 80) out[i] = (i < NCLS - 1) ? bs2[i] : 0.f;
}

// ---------------- attention logits: logits[b,t] = feat[b,t,:]·Watt + b ----------------
__global__ __launch_bounds__(256) void att_logits_kernel(
    const float* __restrict__ feat, const float* __restrict__ Watt,
    const float* __restrict__ batt, float* __restrict__ logits) {
  int row  = blockIdx.x * 8 + (threadIdx.x >> 5);
  int lane = threadIdx.x & 31;
  if (row >= BB * TT) return;
  const float* f = feat + (size_t)row * HH;
  float s = 0.f;
  for (int i = lane; i < HH; i += 32) s += f[i] * Watt[i];
#pragma unroll
  for (int off = 16; off; off >>= 1) s += __shfl_xor(s, off, 32);
  if (lane == 0) logits[row] = s + batt[0];
}

// ---------------- arg span embedding: [startH | endH | widthWD | attH] bf16, padded ----------------
__global__ __launch_bounds__(128) void build_arg_emb_kernel(
    const float* __restrict__ feat, const int* __restrict__ cand,
    const float* __restrict__ logits, const float* __restrict__ Wwidth,
    bf16_t* __restrict__ emb, float* __restrict__ amask) {
  int span = blockIdx.x;            // b*NA + i
  int b    = span / NA;
  int tid  = threadIdx.x;
  int st = cand[span * 2 + 0], en = cand[span * 2 + 1];
  float m = (en > 0) ? 1.f : 0.f;
  int ms = (en > 0) ? st : 0;
  int me = (en > 0) ? en : 0;

  __shared__ float satt[NWW];
  __shared__ float wts[NWW];
  __shared__ int   sidx[NWW];
  if (tid < NWW) {
    int raw = en - tid;
    bool valid = (tid <= en - st) && (raw >= 0);
    int id = raw > 0 ? raw : 0;
    sidx[tid] = id;
    satt[tid] = valid ? logits[b * TT + id] : -1e30f;
  }
  __syncthreads();
  if (tid == 0) {
    float mx = -INFINITY;
    for (int w = 0; w < NWW; ++w) mx = fmaxf(mx, satt[w]);
    float e[NWW]; float sum = 0.f;
    for (int w = 0; w < NWW; ++w) { e[w] = __expf(satt[w] - mx); sum += e[w]; }
    float inv = 1.f / sum;
    for (int w = 0; w < NWW; ++w) wts[w] = (satt[w] > -1e29f) ? e[w] * inv : 0.f;
  }
  __syncthreads();

  const float* fb = feat + (size_t)b * TT * HH;
  size_t ro = (size_t)span * ARG_DP;
  for (int h = tid; h < HH; h += 128) {
    float sf = fb[(size_t)ms * HH + h];
    float ef = fb[(size_t)me * HH + h];
    float at = 0.f;
#pragma unroll
    for (int w = 0; w < NWW; ++w) at += wts[w] * fb[(size_t)sidx[w] * HH + h];
    emb[ro + h]                 = f2bf(sf * m);
    emb[ro + HH + h]            = f2bf(ef * m);
    emb[ro + 2 * HH + WDD + h]  = f2bf(at * m);
  }
  if (tid < WDD) {
    int wd = me - ms; wd = wd < 0 ? 0 : (wd > NWW - 1 ? NWW - 1 : wd);
    emb[ro + 2 * HH + tid] = f2bf(Wwidth[wd * WDD + tid] * m);
  }
  if (tid >= WDD && tid < WDD + (ARG_DP - ARG_D))
    emb[ro + ARG_D + (tid - WDD)] = f2bf(0.f);
  if (tid == 0) amask[span] = m;
}

// ---------------- pred span embedding: [startH | endH] bf16 ----------------
__global__ __launch_bounds__(128) void build_pred_emb_kernel(
    const float* __restrict__ feat, const int* __restrict__ cand,
    bf16_t* __restrict__ emb, float* __restrict__ pmask) {
  int span = blockIdx.x;
  int b    = span / NP;
  int tid  = threadIdx.x;
  int st = cand[span * 2 + 0], en = cand[span * 2 + 1];
  float m = (en > 0) ? 1.f : 0.f;
  int ms = (en > 0) ? st : 0;
  int me = (en > 0) ? en : 0;
  const float* fb = feat + (size_t)b * TT * HH;
  size_t ro = (size_t)span * PRED_D;
  for (int h = tid; h < HH; h += 128) {
    emb[ro + h]      = f2bf(fb[(size_t)ms * HH + h] * m);
    emb[ro + HH + h] = f2bf(fb[(size_t)me * HH + h] * m);
  }
  if (tid == 0) pmask[span] = m;
}

// ---------------- generic bf16 WMMA GEMM: C[M,N] = act(A[M,K] @ Bt[N,K]^T + bias) ----------------
// One wave computes a 16 x (16*NSUB) output tile; NSUB is a compile-time constant so the
// K-loop is straight-line. All fragment loads are issued before any WMMA so the scheduler
// can clause them and stagger s_wait_loadcnt across the NSUB independent WMMA chains.
// A/B fragment layout per CDNA5 ISA 7.12.2: lanes 0-15 hold K{k..k+7, k+16..k+23},
// lanes 16-31 hold K{k+8..k+15, k+24..k+31}, row/col = lane&15.
template <int NSUB>
__global__ __launch_bounds__(32) void gemm_bf16_wmma_kernel(
    const bf16_t* __restrict__ A, int lda,
    const bf16_t* __restrict__ Bt, int ldb,
    float* __restrict__ C, int ldc,
    const float* __restrict__ bias, int K, int relu) {
  const int lane = threadIdx.x;
  const int half = lane >> 4;
  const int l15  = lane & 15;
  const int m0 = blockIdx.y << 4;
  const int n0 = blockIdx.x * (16 * NSUB);

  const bf16_t* ap = A + (size_t)(m0 + l15) * lda + half * 8;
  const bf16_t* bp[NSUB];
#pragma unroll
  for (int s = 0; s < NSUB; ++s)
    bp[s] = Bt + (size_t)(n0 + 16 * s + l15) * ldb + half * 8;

  v8f acc[NSUB] = {};
  for (int k = 0; k < K; k += 32) {
    if (k + 256 < K) {
      __builtin_prefetch(ap + k + 256, 0, 3);      // global_prefetch_b8, high locality
      __builtin_prefetch(bp[0] + k + 256, 0, 3);
    }
    // issue every load first, then the WMMA chain -> staggered s_wait_loadcnt
    v16bf a = load_frag(ap + k);
    v16bf bfrag[NSUB];
#pragma unroll
    for (int s = 0; s < NSUB; ++s) bfrag[s] = load_frag(bp[s] + k);
#pragma unroll
    for (int s = 0; s < NSUB; ++s)
      acc[s] = __builtin_amdgcn_wmma_f32_16x16x32_bf16(
          false, a, false, bfrag[s], (short)0, acc[s], false, false);
  }
  // C/D layout: VGPR v -> M = v (+8 for lanes 16-31), N = lane&15
#pragma unroll
  for (int s = 0; s < NSUB; ++s) {
    int n = n0 + 16 * s + l15;
    float bv = bias ? bias[n] : 0.f;
#pragma unroll
    for (int v = 0; v < 8; ++v) {
      int mrow = m0 + v + half * 8;
      float x = acc[s][v] + bv;
      if (relu) x = fmaxf(x, 0.f);
      C[(size_t)mrow * ldc + n] = x;
    }
  }
}

// ---------------- scores[m] = mask ? hidden[m,:]·W2 + b2 : NEG ----------------
__global__ __launch_bounds__(256) void score_reduce_kernel(
    const float* __restrict__ hidden, const float* __restrict__ W2,
    const float* __restrict__ b2, const float* __restrict__ mask,
    float* __restrict__ ws_scores, float* __restrict__ out_scores, int Mtot) {
  int row  = blockIdx.x * 8 + (threadIdx.x >> 5);
  int lane = threadIdx.x & 31;
  if (row >= Mtot) return;
  const float* h = hidden + (size_t)row * HH;
  float s = 0.f;
  for (int i = lane; i < HH; i += 32) s += h[i] * W2[i];
#pragma unroll
  for (int off = 16; off; off >>= 1) s += __shfl_xor(s, off, 32);
  if (lane == 0) {
    float sc = (mask[row] > 0.f) ? (s + b2[0]) : NEGV;
    ws_scores[row]  = sc;
    out_scores[row] = sc;
  }
}

// ---------------- per-batch iterative top-K + fill/sort (one wave per batch) ----------------
__global__ __launch_bounds__(32) void topk_kernel(
    const float* __restrict__ scores, const float* __restrict__ mask,
    const int* __restrict__ cand, int N, int K, float ratio,
    int* __restrict__ top_idx, float* __restrict__ top_scores,
    float* __restrict__ out_spans, float* __restrict__ out_mask) {
  __shared__ float sc[1024];
  __shared__ int   sel[32];
  __shared__ int   snk;
  int b = blockIdx.x, lane = threadIdx.x;
  float msum = 0.f;
  for (int n = lane; n < N; n += 32) { sc[n] = scores[b * N + n]; msum += mask[b * N + n]; }
#pragma unroll
  for (int off = 16; off; off >>= 1) msum += __shfl_xor(msum, off, 32);
  __syncthreads();
  for (int k = 0; k < K; ++k) {
    float bv = -INFINITY; int bi = N;
    for (int n = lane; n < N; n += 32) {
      float v = sc[n];
      if (v > bv) { bv = v; bi = n; }   // increasing n keeps lowest index on ties
    }
#pragma unroll
    for (int off = 16; off; off >>= 1) {
      float ov = __shfl_xor(bv, off, 32);
      int   oi = __shfl_xor(bi, off, 32);
      if (ov > bv || (ov == bv && oi < bi)) { bv = ov; bi = oi; }
    }
    if (lane == 0) { sel[k] = bi; sc[bi] = -INFINITY; }
    __syncthreads();
  }
  if (lane == 0) {
    int nk = (int)(msum * ratio); if (nk > K) nk = K;
    int fill = 0;
    for (int k = 0; k < K; ++k) if (sel[k] > fill) fill = sel[k];
    for (int k = nk; k < K; ++k) sel[k] = fill;
    for (int i = 1; i < K; ++i) {       // insertion sort ascending
      int v = sel[i], j = i - 1;
      while (j >= 0 && sel[j] > v) { sel[j + 1] = sel[j]; --j; }
      sel[j + 1] = v;
    }
    snk = nk;
  }
  __syncthreads();
  if (lane < K) {
    int idx = sel[lane];
    top_idx[b * K + lane]    = idx;
    top_scores[b * K + lane] = scores[b * N + idx];
    out_mask[b * K + lane]   = (lane < snk && mask[b * N + idx] > 0.f) ? 1.f : 0.f;
    out_spans[(b * K + lane) * 2 + 0] = (float)cand[(b * N + idx) * 2 + 0];
    out_spans[(b * K + lane) * 2 + 1] = (float)cand[(b * N + idx) * 2 + 1];
  }
}

// ---------------- gather top-k embedding rows into padded compact A matrix ----------------
__global__ void gather_rows_kernel(const bf16_t* __restrict__ emb, int ld,
                                   const int* __restrict__ idx, int K, int Kpad,
                                   int NB, bf16_t* __restrict__ out) {
  size_t gid = (size_t)blockIdx.x * blockDim.x + threadIdx.x;
  size_t tot = (size_t)BB * Kpad * ld;
  if (gid >= tot) return;
  int c = (int)(gid % ld);
  size_t r = gid / ld;
  int k = (int)(r % Kpad);
  int b = (int)(r / Kpad);
  bf16_t v = f2bf(0.f);
  if (k < K) {
    int id = idx[b * K + k];
    v = emb[((size_t)(b * NB + id)) * ld + c];
  }
  out[gid] = v;
}

// ---------------- h = relu(pred_proj + arg_proj + bs1) -> bf16 rows [B*Kp*Ka, H] ----------------
__global__ void build_h_kernel(const float* __restrict__ pp, const float* __restrict__ ap,
                               const float* __restrict__ bs1, bf16_t* __restrict__ h) {
  size_t gid = (size_t)blockIdx.x * blockDim.x + threadIdx.x;
  size_t tot = (size_t)BB * KPRED * KARG * HH;
  if (gid >= tot) return;
  int n = (int)(gid & (HH - 1));
  size_t row = gid >> 10;
  int a  = (int)(row % KARG);
  size_t r2 = row / KARG;
  int p  = (int)(r2 % KPRED);
  int b  = (int)(r2 / KPRED);
  float v = pp[((size_t)(b * 16 + p)) * HH + n] + ap[((size_t)(b * 32 + a)) * HH + n] + bs1[n];
  h[gid] = f2bf(fmaxf(v, 0.f));
}

// ---------------- srl_scores assembly: dummy class 0, add pruner scores ----------------
__global__ void epilogue_kernel(const float* __restrict__ pairC,
                                const float* __restrict__ ats, const float* __restrict__ pts,
                                float* __restrict__ srl) {
  size_t gid = (size_t)blockIdx.x * blockDim.x + threadIdx.x;
  size_t tot = (size_t)BB * KPRED * KARG * NCLS;
  if (gid >= tot) return;
  int c = (int)(gid % NCLS);
  size_t r = gid / NCLS;               // (b*Kp + p)*Ka + a
  int a = (int)(r % KARG);
  size_t r2 = r / KARG;
  int p = (int)(r2 % KPRED);
  int b = (int)(r2 / KPRED);
  float v = 0.f;
  if (c > 0) v = pairC[r * 80 + (c - 1)] + ats[b * KARG + a] + pts[b * KPRED + p];
  srl[gid] = v;
}

// ==================== host orchestration ====================
static inline int cdiv(size_t a, size_t b) { return (int)((a + b - 1) / b); }

extern "C" void kernel_launch(void* const* d_in, const int* in_sizes, int n_in,
                              void* d_out, int out_size, void* d_ws, size_t ws_size,
                              hipStream_t stream) {
  const float* feat   = (const float*)d_in[0];
  const int*   acand  = (const int*)d_in[1];
  const int*   pcand  = (const int*)d_in[2];
  const float* Wwidth = (const float*)d_in[3];
  const float* Watt   = (const float*)d_in[4];
  const float* batt   = (const float*)d_in[5];
  const float* Wa1    = (const float*)d_in[6];
  const float* ba1    = (const float*)d_in[7];
  const float* Wa2    = (const float*)d_in[8];
  const float* ba2    = (const float*)d_in[9];
  const float* Wp1    = (const float*)d_in[10];
  const float* bp1    = (const float*)d_in[11];
  const float* Wp2    = (const float*)d_in[12];
  const float* bp2    = (const float*)d_in[13];
  const float* Ws1    = (const float*)d_in[14];
  const float* bs1    = (const float*)d_in[15];
  const float* Ws2    = (const float*)d_in[16];
  const float* bs2    = (const float*)d_in[17];
  float* out = (float*)d_out;

  // output regions (reference return order, flattened)
  float* o_srl = out;                                  // [B,15,30,66]
  float* o_tps = o_srl + (size_t)BB * KPRED * KARG * NCLS;
  float* o_tas = o_tps + (size_t)BB * KPRED * 2;
  float* o_tpm = o_tas + (size_t)BB * KARG * 2;
  float* o_tam = o_tpm + (size_t)BB * KPRED;
  float* o_pfs = o_tam + (size_t)BB * KARG;
  float* o_afs = o_pfs + (size_t)BB * NP;

  // workspace bump allocator (256B aligned)
  char* base = (char*)d_ws;
  size_t off = 0;
  auto alloc = [&](size_t bytes) -> void* {
    void* r = base + off;
    off = (off + bytes + 255) & ~(size_t)255;
    return r;
  };
  float*  logits  = (float*)alloc((size_t)BB * TT * 4);
  bf16_t* argEmb  = (bf16_t*)alloc((size_t)BB * NA * ARG_DP * 2);
  bf16_t* predEmb = (bf16_t*)alloc((size_t)BB * NP * PRED_D * 2);
  bf16_t* Wa1t    = (bf16_t*)alloc((size_t)HH * ARG_DP * 2);
  bf16_t* Wp1t    = (bf16_t*)alloc((size_t)HH * PRED_D * 2);
  bf16_t* Wspt    = (bf16_t*)alloc((size_t)HH * PRED_D * 2);
  bf16_t* Wsat    = (bf16_t*)alloc((size_t)HH * ARG_DP * 2);
  bf16_t* Ws2t    = (bf16_t*)alloc((size_t)80 * HH * 2);
  float*  bias80  = (float*)alloc(80 * 4);
  float*  hidA    = (float*)alloc((size_t)BB * NA * HH * 4);
  float*  hidP    = (float*)alloc((size_t)BB * NP * HH * 4);
  float*  amask   = (float*)alloc((size_t)BB * NA * 4);
  float*  pmask   = (float*)alloc((size_t)BB * NP * 4);
  float*  ascore  = (float*)alloc((size_t)BB * NA * 4);
  float*  pscore  = (float*)alloc((size_t)BB * NP * 4);
  int*    aidx    = (int*)alloc((size_t)BB * KARG * 4);
  int*    pidx    = (int*)alloc((size_t)BB * KPRED * 4);
  float*  ats     = (float*)alloc((size_t)BB * KARG * 4);
  float*  pts     = (float*)alloc((size_t)BB * KPRED * 4);
  bf16_t* topA    = (bf16_t*)alloc((size_t)BB * 32 * ARG_DP * 2);
  bf16_t* topP    = (bf16_t*)alloc((size_t)BB * 16 * PRED_D * 2);
  float*  pproj   = (float*)alloc((size_t)BB * 16 * HH * 4);
  float*  aproj   = (float*)alloc((size_t)BB * 32 * HH * 4);
  bf16_t* hbuf    = (bf16_t*)alloc((size_t)BB * KPRED * KARG * HH * 2);
  float*  pairC   = (float*)alloc((size_t)BB * KPRED * KARG * 80 * 4);
  (void)in_sizes; (void)n_in; (void)out_size; (void)ws_size;

  // 1) weight prep (transpose + bf16)
  transpose_cvt_kernel<<<cdiv((size_t)HH * ARG_DP, 256), 256, 0, stream>>>(Wa1, ARG_D, HH, Wa1t, ARG_DP, HH);
  transpose_cvt_kernel<<<cdiv((size_t)HH * PRED_D, 256), 256, 0, stream>>>(Wp1, PRED_D, HH, Wp1t, PRED_D, HH);
  transpose_cvt_kernel<<<cdiv((size_t)HH * PRED_D, 256), 256, 0, stream>>>(Ws1, PRED_D, HH, Wspt, PRED_D, HH);
  transpose_cvt_kernel<<<cdiv((size_t)HH * ARG_DP, 256), 256, 0, stream>>>(Ws1 + (size_t)PRED_D * HH, ARG_D, HH, Wsat, ARG_DP, HH);
  transpose_cvt_kernel<<<cdiv((size_t)80 * HH, 256), 256, 0, stream>>>(Ws2, HH, NCLS - 1, Ws2t, HH, 80);
  pad_bias_kernel<<<1, 128, 0, stream>>>(bs2, bias80);

  // 2) global attention logits
  att_logits_kernel<<<(BB * TT) / 8, 256, 0, stream>>>(feat, Watt, batt, logits);

  // 3) span embeddings (bf16, K-padded)
  build_arg_emb_kernel<<<BB * NA, 128, 0, stream>>>(feat, acand, logits, Wwidth, argEmb, amask);
  build_pred_emb_kernel<<<BB * NP, 128, 0, stream>>>(feat, pcand, predEmb, pmask);

  // 4) pruning MLP hidden layers (WMMA GEMMs, relu+bias fused); N=1024 -> NSUB=4, grid.x=16
  {
    dim3 g(HH / 64, (BB * NA) / 16);
    gemm_bf16_wmma_kernel<4><<<g, 32, 0, stream>>>(argEmb, ARG_DP, Wa1t, ARG_DP, hidA, HH, ba1, ARG_DP, 1);
  }
  {
    dim3 g(HH / 64, (BB * NP) / 16);
    gemm_bf16_wmma_kernel<4><<<g, 32, 0, stream>>>(predEmb, PRED_D, Wp1t, PRED_D, hidP, HH, bp1, PRED_D, 1);
  }

  // 5) pruning scores (+ full-score outputs)
  score_reduce_kernel<<<cdiv(BB * NA, 8), 256, 0, stream>>>(hidA, Wa2, ba2, amask, ascore, o_afs, BB * NA);
  score_reduce_kernel<<<cdiv(BB * NP, 8), 256, 0, stream>>>(hidP, Wp2, bp2, pmask, pscore, o_pfs, BB * NP);

  // 6) top-k prune (writes spans + masks to output)
  topk_kernel<<<BB, 32, 0, stream>>>(ascore, amask, acand, NA, KARG, 0.8f, aidx, ats, o_tas, o_tam);
  topk_kernel<<<BB, 32, 0, stream>>>(pscore, pmask, pcand, NP, KPRED, 0.4f, pidx, pts, o_tps, o_tpm);

  // 7) gather top embeddings into compact padded matrices
  gather_rows_kernel<<<cdiv((size_t)BB * 32 * ARG_DP, 256), 256, 0, stream>>>(argEmb, ARG_DP, aidx, KARG, 32, NA, topA);
  gather_rows_kernel<<<cdiv((size_t)BB * 16 * PRED_D, 256), 256, 0, stream>>>(predEmb, PRED_D, pidx, KPRED, 16, NP, topP);

  // 8) pairwise projections (WMMA)
  {
    dim3 g(HH / 64, (BB * 16) / 16);
    gemm_bf16_wmma_kernel<4><<<g, 32, 0, stream>>>(topP, PRED_D, Wspt, PRED_D, pproj, HH, nullptr, PRED_D, 0);
  }
  {
    dim3 g(HH / 64, (BB * 32) / 16);
    gemm_bf16_wmma_kernel<4><<<g, 32, 0, stream>>>(topA, ARG_DP, Wsat, ARG_DP, aproj, HH, nullptr, ARG_DP, 0);
  }

  // 9) h = relu(pp + ap + bs1), then classifier GEMM h @ Ws2:
  //    M = 3600 = 225*16 tiles, N = 80 exactly -> NSUB=5, grid.x=1 (no tail logic)
  build_h_kernel<<<cdiv((size_t)BB * KPRED * KARG * HH, 256), 256, 0, stream>>>(pproj, aproj, bs1, hbuf);
  {
    dim3 g(1, (BB * KPRED * KARG) / 16);
    gemm_bf16_wmma_kernel<5><<<g, 32, 0, stream>>>(hbuf, HH, Ws2t, HH, pairC, 80, bias80, HH, 0);
  }

  // 10) assemble srl_scores with dummy class 0 and pruner-score broadcasts
  epilogue_kernel<<<cdiv((size_t)BB * KPRED * KARG * NCLS, 256), 256, 0, stream>>>(pairC, ats, pts, o_srl);
}
